// AllSet_81020263071820
// MI455X (gfx1250) — compile-verified, matrix operand order checked by reading, not run on hardware
//
#include <hip/hip_runtime.h>
#include <math.h>

// Problem constants (match reference)
#define NN  100000
#define MM  800000
#define EHH 50000
#define DDIM 128

typedef __attribute__((ext_vector_type(16))) _Float16 v16h;
typedef __attribute__((ext_vector_type(8)))  _Float16 v8h;
typedef __attribute__((ext_vector_type(8)))  float    v8f;

// ---------------------------------------------------------------------------
// Weight prep: transpose 128x128 f32 weight -> f16 Wt[n][k] (column access is
// then contiguous per output column, matching the WMMA B-fragment layout).
// ---------------------------------------------------------------------------
__global__ void k_w_to_f16t(const float* __restrict__ W, _Float16* __restrict__ Wt) {
    int t = blockIdx.x * blockDim.x + threadIdx.x;   // 16384 threads
    if (t < 128 * 128) {
        int n = t & 127, k = t >> 7;
        Wt[n * 128 + k] = (_Float16)W[k * 128 + n];
    }
}

// Collapse xK-projection: Kvec[d][h] = sum_c Kw[d, h*32+c]*att[h,c];
// kconst[h] = sum_c Kb[h*32+c]*att[h,c]
__global__ void k_kvec(const float* __restrict__ Kw, const float* __restrict__ Kb,
                       const float* __restrict__ att,
                       float* __restrict__ Kvec, float* __restrict__ kconst) {
    int t = blockIdx.x * blockDim.x + threadIdx.x;
    if (t < 512) {
        int d = t >> 2, h = t & 3;
        float s = 0.f;
        #pragma unroll
        for (int c = 0; c < 32; c++) s += Kw[d * 128 + h * 32 + c] * att[h * 32 + c];
        Kvec[d * 4 + h] = s;
    }
    if (t < 4) {
        float s = 0.f;
        #pragma unroll
        for (int c = 0; c < 32; c++) s += Kb[t * 32 + c] * att[t * 32 + c];
        kconst[t] = s;
    }
}

__global__ void k_fill(unsigned* __restrict__ p, unsigned v, int n) {
    int t = blockIdx.x * blockDim.x + threadIdx.x;
    if (t < n) p[t] = v;
}

// ---------------------------------------------------------------------------
// WMMA GEMM: out[rows,128] = op(A[rows,128] @ W + bias), W given as f16
// transposed Wt[n][k]. Block = 256 threads = 8 waves; each wave owns one
// 16x16 output tile of a 16-row strip. Epilogue staged through LDS so the
// final stores are row-contiguous b128.
// ---------------------------------------------------------------------------
__global__ void k_gemm128(const float* __restrict__ A, const _Float16* __restrict__ Wt,
                          const float* __restrict__ bias, float* __restrict__ out,
                          int rows, int do_relu) {
    __shared__ __align__(16) char smem[16 * 132 * 4];   // 8448 B, reused A->D
    _Float16* As = (_Float16*)smem;                      // stride 136 f16/row
    float*    Ds = (float*)smem;                         // stride 132 f32/row

    const int tid = threadIdx.x;
    const int r0  = blockIdx.x * 16;

    // Cooperative load of the 16x128 A strip, converted to f16.
    #pragma unroll
    for (int j = 0; j < 8; j++) {
        int idx = tid + 256 * j;                 // 2048 elements
        int rr = idx >> 7, cc = idx & 127;
        float v = (r0 + rr < rows) ? A[(long)(r0 + rr) * 128 + cc] : 0.f;
        As[rr * 136 + cc] = (_Float16)v;
    }
    __syncthreads();

    const int wave = tid >> 5;
    const int lane = tid & 31;
    const int n0   = wave * 16;
    const int rowA = lane & 15;
    const int kb   = (lane & 16) ? 8 : 0;        // K sub-block per lane half
    const int nB   = n0 + (lane & 15);

    v8f acc = {};
    #pragma unroll
    for (int kk = 0; kk < 4; kk++) {             // K = 4 x 32
        int kbase = kk * 32 + kb;
        v8h alo = *(const v8h*)&As[rowA * 136 + kbase];
        v8h ahi = *(const v8h*)&As[rowA * 136 + kbase + 16];
        v8h blo = *(const v8h*)(Wt + nB * 128 + kbase);
        v8h bhi = *(const v8h*)(Wt + nB * 128 + kbase + 16);
        v16h a, b;
        #pragma unroll
        for (int e = 0; e < 8; e++) { a[e] = alo[e]; a[e + 8] = ahi[e];
                                      b[e] = blo[e]; b[e + 8] = bhi[e]; }
        acc = __builtin_amdgcn_wmma_f32_16x16x32_f16(
                  false, a, false, b, (short)0, acc, false, false);
    }
    __syncthreads();                              // A staging no longer needed

    // C/D layout: element i of lane -> (m = (lane&16?8:0)+i, n = n0+(lane&15))
    const int mBase = (lane & 16) ? 8 : 0;
    const int n = n0 + (lane & 15);
    const float bn = bias[n];
    #pragma unroll
    for (int i = 0; i < 8; i++) {
        float val = acc[i] + bn;
        if (do_relu) val = fmaxf(val, 0.f);
        Ds[(mBase + i) * 132 + n] = val;
    }
    __syncthreads();

    // Coalesced b128 stores: 512 float4 chunks, 2 per thread.
    #pragma unroll
    for (int j = 0; j < 2; j++) {
        int idx = tid + 256 * j;                 // chunk id
        int rr = idx >> 5, cc4 = (idx & 31) * 4;
        if (r0 + rr < rows) {
            float4 v = { Ds[rr * 132 + cc4 + 0], Ds[rr * 132 + cc4 + 1],
                         Ds[rr * 132 + cc4 + 2], Ds[rr * 132 + cc4 + 3] };
            *(float4*)&out[(long)(r0 + rr) * 128 + cc4] = v;
        }
    }
}

// ---------------------------------------------------------------------------
// alpha_lr[r,h] = leaky_relu( x[r,:] @ Kvec[:,h] + kconst[h], 0.2 )
// One wave32 per row; float4 loads, shuffle reduction, all 4 heads at once.
// ---------------------------------------------------------------------------
__global__ void k_alpha(const float* __restrict__ x, const float* __restrict__ Kvec,
                        const float* __restrict__ kconst, float* __restrict__ alpha,
                        int rows) {
    int wave = threadIdx.x >> 5, lane = threadIdx.x & 31;
    int r = blockIdx.x * 8 + wave;
    if (r >= rows) return;
    const float4 xv = *(const float4*)&x[(long)r * 128 + lane * 4];
    const float xj[4] = { xv.x, xv.y, xv.z, xv.w };
    float ph[4] = { 0.f, 0.f, 0.f, 0.f };
    #pragma unroll
    for (int j = 0; j < 4; j++) {
        const float4 kv = *(const float4*)&Kvec[(lane * 4 + j) * 4];
        ph[0] += xj[j] * kv.x; ph[1] += xj[j] * kv.y;
        ph[2] += xj[j] * kv.z; ph[3] += xj[j] * kv.w;
    }
    #pragma unroll
    for (int off = 16; off; off >>= 1) {
        #pragma unroll
        for (int h = 0; h < 4; h++) ph[h] += __shfl_xor(ph[h], off, 32);
    }
    if (lane == 0) {
        float4 o;
        float a0 = ph[0] + kconst[0]; o.x = a0 > 0.f ? a0 : 0.2f * a0;
        float a1 = ph[1] + kconst[1]; o.y = a1 > 0.f ? a1 : 0.2f * a1;
        float a2 = ph[2] + kconst[2]; o.z = a2 > 0.f ? a2 : 0.2f * a2;
        float a3 = ph[3] + kconst[3]; o.w = a3 > 0.f ? a3 : 0.2f * a3;
        *(float4*)&alpha[r * 4] = o;
    }
}

// monotone float<->uint encoding for atomicMax over signed floats
__device__ __forceinline__ unsigned enc_f(float f) {
    unsigned u = __float_as_uint(f);
    return (u & 0x80000000u) ? ~u : (u | 0x80000000u);
}
__device__ __forceinline__ float dec_f(unsigned u) {
    return (u & 0x80000000u) ? __uint_as_float(u & 0x7fffffffu) : __uint_as_float(~u);
}
#define ENC_NEG_INF 0x007FFFFFu   // enc(-inf)

__global__ void k_segmax(const int* __restrict__ src, const int* __restrict__ dst,
                         int m, int anc_s, int anc_d,
                         const float* __restrict__ alpha, unsigned* __restrict__ amax) {
    int t = blockIdx.x * blockDim.x + threadIdx.x;
    if (t >= (m + 1) * 4) return;
    int i = t >> 2, h = t & 3;
    int s = (i < m) ? src[i] : anc_s;
    int d = (i < m) ? dst[i] : anc_d;
    atomicMax(&amax[d * 4 + h], enc_f(alpha[s * 4 + h]));
}

__global__ void k_expsum(const int* __restrict__ src, const int* __restrict__ dst,
                         int m, int anc_s, int anc_d,
                         const float* __restrict__ alpha, const unsigned* __restrict__ amax,
                         float* __restrict__ e, float* __restrict__ denom) {
    int t = blockIdx.x * blockDim.x + threadIdx.x;
    if (t >= (m + 1) * 4) return;
    int i = t >> 2, h = t & 3;
    int s = (i < m) ? src[i] : anc_s;
    int d = (i < m) ? dst[i] : anc_d;
    float ev = expf(alpha[s * 4 + h] - dec_f(amax[d * 4 + h]));
    e[i * 4 + h] = ev;
    atomicAdd(&denom[d * 4 + h], ev);
}

// Wave-per-incidence scatter: each lane owns 4 contiguous columns (same head),
// one b128 gather of xV, 4 L2-resident f32 atomic adds.
__global__ void k_scatter(const int* __restrict__ src, const int* __restrict__ dst,
                          int m, int anc_s, int anc_d,
                          const float* __restrict__ e, const float* __restrict__ denom,
                          const float* __restrict__ xV, float* __restrict__ acc) {
    int wave = threadIdx.x >> 5, lane = threadIdx.x & 31;
    int i = blockIdx.x * 8 + wave;
    if (i > m) return;
    int s = (i < m) ? src[i] : anc_s;
    int d = (i < m) ? dst[i] : anc_d;
    int c0 = lane * 4, h = lane >> 3;
    float al = e[i * 4 + h] / (denom[d * 4 + h] + 1e-16f);
    const float4 xv = *(const float4*)&xV[(long)s * 128 + c0];
    float* ap = &acc[(long)d * 128 + c0];
    atomicAdd(ap + 0, xv.x * al);
    atomicAdd(ap + 1, xv.y * al);
    atomicAdd(ap + 2, xv.z * al);
    atomicAdd(ap + 3, xv.w * al);
}

// ---------------------------------------------------------------------------
// LayerNorm kernels: one wave32 per row, float4 per lane, shuffle reductions.
// ---------------------------------------------------------------------------
__global__ void k_ln_bias(const float* __restrict__ in, const float* __restrict__ attb,
                          const float* __restrict__ s, const float* __restrict__ b,
                          float* __restrict__ out, int rows) {
    int wave = threadIdx.x >> 5, lane = threadIdx.x & 31;
    int r = blockIdx.x * 8 + wave;
    if (r >= rows) return;
    const float4 xin = *(const float4*)&in[(long)r * 128 + lane * 4];
    const float4 ab  = *(const float4*)&attb[lane * 4];
    float v[4] = { xin.x + ab.x, xin.y + ab.y, xin.z + ab.z, xin.w + ab.w };
    float sm = 0.f, sq = 0.f;
    #pragma unroll
    for (int j = 0; j < 4; j++) { sm += v[j]; sq += v[j] * v[j]; }
    #pragma unroll
    for (int off = 16; off; off >>= 1) { sm += __shfl_xor(sm, off, 32); sq += __shfl_xor(sq, off, 32); }
    float mean = sm * (1.f / 128.f);
    float var  = sq * (1.f / 128.f) - mean * mean;
    float rstd = rsqrtf(var + 1e-5f);
    const float4 s4 = *(const float4*)&s[lane * 4];
    const float4 b4 = *(const float4*)&b[lane * 4];
    float4 o = { (v[0] - mean) * rstd * s4.x + b4.x, (v[1] - mean) * rstd * s4.y + b4.y,
                 (v[2] - mean) * rstd * s4.z + b4.z, (v[3] - mean) * rstd * s4.w + b4.w };
    *(float4*)&out[(long)r * 128 + lane * 4] = o;
}

__global__ void k_ln_res(const float* __restrict__ y0, const float* __restrict__ ff,
                         const float* __restrict__ s, const float* __restrict__ b,
                         float* __restrict__ out, int rows, int do_relu) {
    int wave = threadIdx.x >> 5, lane = threadIdx.x & 31;
    int r = blockIdx.x * 8 + wave;
    if (r >= rows) return;
    const float4 ya = *(const float4*)&y0[(long)r * 128 + lane * 4];
    const float4 fa = *(const float4*)&ff[(long)r * 128 + lane * 4];
    float v[4] = { ya.x + fa.x, ya.y + fa.y, ya.z + fa.z, ya.w + fa.w };
    float sm = 0.f, sq = 0.f;
    #pragma unroll
    for (int j = 0; j < 4; j++) { sm += v[j]; sq += v[j] * v[j]; }
    #pragma unroll
    for (int off = 16; off; off >>= 1) { sm += __shfl_xor(sm, off, 32); sq += __shfl_xor(sq, off, 32); }
    float mean = sm * (1.f / 128.f);
    float var  = sq * (1.f / 128.f) - mean * mean;
    float rstd = rsqrtf(var + 1e-5f);
    const float4 s4 = *(const float4*)&s[lane * 4];
    const float4 b4 = *(const float4*)&b[lane * 4];
    float o[4] = { (v[0] - mean) * rstd * s4.x + b4.x, (v[1] - mean) * rstd * s4.y + b4.y,
                   (v[2] - mean) * rstd * s4.z + b4.z, (v[3] - mean) * rstd * s4.w + b4.w };
    if (do_relu) {
        #pragma unroll
        for (int j = 0; j < 4; j++) o[j] = fmaxf(o[j], 0.f);
    }
    float4 ov = { o[0], o[1], o[2], o[3] };
    *(float4*)&out[(long)r * 128 + lane * 4] = ov;
}

// ---------------------------------------------------------------------------
// Host-side PMA pass orchestration
// ---------------------------------------------------------------------------
static inline unsigned cdiv(long a, long b) { return (unsigned)((a + b - 1) / b); }

static void run_pma(const float* x, int n_src, const int* srcIdx, const int* dstIdx,
                    int anc_s, int anc_d, int n_dst,
                    const _Float16* Vwt, const float* Vb,
                    const float* Kvec, const float* kconst, const float* attb,
                    const _Float16* w1t, const float* b1,
                    const _Float16* w2t, const float* b2,
                    const float* ln0s, const float* ln0b,
                    const float* ln1s, const float* ln1b,
                    float* xV, float* alphaB, float* eB, unsigned* amaxB,
                    float* denomB, float* accB, float* y0B, float* h1B,
                    float* outPtr, int relu_out, hipStream_t stream) {
    // projections
    k_gemm128<<<cdiv(n_src, 16), 256, 0, stream>>>(x, Vwt, Vb, xV, n_src, 0);
    k_alpha  <<<cdiv(n_src, 8), 256, 0, stream>>>(x, Kvec, kconst, alphaB, n_src);
    // init segment buffers
    k_fill<<<cdiv((long)n_dst * 4, 256), 256, 0, stream>>>(amaxB, ENC_NEG_INF, n_dst * 4);
    k_fill<<<cdiv((long)n_dst * 4, 256), 256, 0, stream>>>((unsigned*)denomB, 0u, n_dst * 4);
    k_fill<<<cdiv((long)n_dst * 128, 256), 256, 0, stream>>>((unsigned*)accB, 0u, n_dst * 128);
    // segment softmax + weighted scatter
    k_segmax <<<cdiv((long)(MM + 1) * 4, 256), 256, 0, stream>>>(srcIdx, dstIdx, MM, anc_s, anc_d, alphaB, amaxB);
    k_expsum <<<cdiv((long)(MM + 1) * 4, 256), 256, 0, stream>>>(srcIdx, dstIdx, MM, anc_s, anc_d, alphaB, amaxB, eB, denomB);
    k_scatter<<<cdiv((long)(MM + 1), 8), 256, 0, stream>>>(srcIdx, dstIdx, MM, anc_s, anc_d, eB, denomB, xV, accB);
    // epilogue: LN0 -> rFF (two WMMA GEMMs) -> residual LN1
    k_ln_bias<<<cdiv(n_dst, 8), 256, 0, stream>>>(accB, attb, ln0s, ln0b, y0B, n_dst);
    k_gemm128<<<cdiv(n_dst, 16), 256, 0, stream>>>(y0B, w1t, b1, h1B, n_dst, 1);
    k_gemm128<<<cdiv(n_dst, 16), 256, 0, stream>>>(h1B, w2t, b2, accB, n_dst, 1);  // reuse acc as ff
    k_ln_res <<<cdiv(n_dst, 8), 256, 0, stream>>>(y0B, accB, ln1s, ln1b, outPtr, n_dst, relu_out);
}

extern "C" void kernel_launch(void* const* d_in, const int* in_sizes, int n_in,
                              void* d_out, int out_size, void* d_ws, size_t ws_size,
                              hipStream_t stream) {
    const float* X      = (const float*)d_in[0];
    const int*   vertex = (const int*)d_in[1];
    const int*   edges  = (const int*)d_in[2];
    // per-pass params: Kw,Kb,Vw,Vb,att,w1,b1,w2,b2,ln0s,ln0b,ln1s,ln1b
    const float* P[26];
    for (int i = 0; i < 26; i++) P[i] = (const float*)d_in[3 + i];

    char* ws = (char*)d_ws;
    auto alloc = [&](size_t bytes) -> char* {
        char* p = ws; ws += (bytes + 255) & ~(size_t)255; return p;
    };
    _Float16 *Vwt[2], *w1t[2], *w2t[2];
    float *Kvec[2], *kcst[2];
    for (int p = 0; p < 2; p++) {
        Vwt[p] = (_Float16*)alloc(128 * 128 * 2);
        w1t[p] = (_Float16*)alloc(128 * 128 * 2);
        w2t[p] = (_Float16*)alloc(128 * 128 * 2);
        Kvec[p] = (float*)alloc(512 * 4);
        kcst[p] = (float*)alloc(256);
    }
    float*    xV     = (float*)   alloc((size_t)(NN + 1) * 128 * 4);
    float*    alphaB = (float*)   alloc((size_t)(NN + 1) * 4 * 4);
    float*    eB     = (float*)   alloc((size_t)(MM + 1) * 4 * 4);
    unsigned* amaxB  = (unsigned*)alloc((size_t)NN * 4 * 4);
    float*    denomB = (float*)   alloc((size_t)NN * 4 * 4);
    float*    accB   = (float*)   alloc((size_t)NN * 128 * 4);
    float*    y0B    = (float*)   alloc((size_t)NN * 128 * 4);
    float*    h1B    = (float*)   alloc((size_t)NN * 128 * 4);
    float*    X1     = (float*)   alloc((size_t)(EHH + 1) * 128 * 4);

    // weight prep (f16 transposed weights + collapsed K-projection vectors)
    for (int p = 0; p < 2; p++) {
        const float* const* B = &P[p * 13];
        k_w_to_f16t<<<64, 256, 0, stream>>>(B[2], Vwt[p]);  // Vw
        k_w_to_f16t<<<64, 256, 0, stream>>>(B[5], w1t[p]);  // w1
        k_w_to_f16t<<<64, 256, 0, stream>>>(B[7], w2t[p]);  // w2
        k_kvec<<<2, 256, 0, stream>>>(B[0], B[1], B[4], Kvec[p], kcst[p]);
    }

    // Pass 1 (V2E): src = vertex (+anchor N-1), dst = edges (+anchor EH), n_dst = EH+1
    {
        const float* const* B = &P[0];
        run_pma(X, NN, vertex, edges, NN - 1, EHH, EHH + 1,
                Vwt[0], B[3], Kvec[0], kcst[0], B[4], w1t[0], B[6], w2t[0], B[8],
                B[9], B[10], B[11], B[12],
                xV, alphaB, eB, amaxB, denomB, accB, y0B, h1B,
                X1, /*relu_out=*/1, stream);
    }
    // Pass 2 (E2V): src = edges (+anchor EH), dst = vertex (+anchor N-1), n_dst = N
    {
        const float* const* B = &P[13];
        run_pma(X1, EHH + 1, edges, vertex, EHH, NN - 1, NN,
                Vwt[1], B[3], Kvec[1], kcst[1], B[4], w1t[1], B[6], w2t[1], B[8],
                B[9], B[10], B[11], B[12],
                xV, alphaB, eB, amaxB, denomB, accB, y0B, h1B,
                (float*)d_out, /*relu_out=*/0, stream);
    }
}
